// BasicTransformerBlock_3667902071065
// MI455X (gfx1250) — matrix operand order repaired and checked by reading, not compiled, not run on hardware
//
#include <hip/hip_runtime.h>
#include <cstdint>

// ---------------------------------------------------------------------------
// Types
// ---------------------------------------------------------------------------
typedef __bf16 bf16_t;
typedef __bf16 bf16x8 __attribute__((ext_vector_type(8)));
typedef __bf16 v16bf  __attribute__((ext_vector_type(16)));
typedef float  v8f    __attribute__((ext_vector_type(8)));

// CDNA5 async global->LDS copies (ASYNCcnt path). Guarded so both toolchains
// compile; fallback is a synchronous VGPR round-trip with identical semantics
// (addresses are always clamped to valid memory by the callers).
#if __has_builtin(__builtin_amdgcn_global_load_async_to_lds_b128) && \
    __has_builtin(__builtin_amdgcn_s_wait_asynccnt)
#define USE_ASYNC_LDS 1
// exact parameter types of the builtin: (global v4i*, local v4i*, imm, imm)
typedef int v4i_vs __attribute__((vector_size(16)));
typedef __attribute__((address_space(1))) v4i_vs* g4p_t;
typedef __attribute__((address_space(3))) v4i_vs* l4p_t;
#else
#define USE_ASYNC_LDS 0
#endif

__device__ __forceinline__ bf16_t f2bf(float f) {
  union { float f; unsigned u; } in; in.f = f;
  unsigned u = in.u;
  u = u + 0x7FFFu + ((u >> 16) & 1u);      // round-to-nearest-even
  unsigned short h = (unsigned short)(u >> 16);
  bf16_t out; __builtin_memcpy(&out, &h, 2);
  return out;
}
__device__ __forceinline__ float bf2f(bf16_t b) {
  unsigned short h; __builtin_memcpy(&h, &b, 2);
  union { unsigned u; float f; } out; out.u = ((unsigned)h) << 16;
  return out.f;
}

// 16-byte global -> LDS copy; async (no VGPR round trip) when available.
__device__ __forceinline__ void async_copy16(const bf16_t* g, bf16_t* l) {
#if USE_ASYNC_LDS
  __builtin_amdgcn_global_load_async_to_lds_b128(
      (g4p_t)(uintptr_t)g,
      (l4p_t)(uint32_t)(uintptr_t)l,
      0, 0);
#else
  *(bf16x8*)l = *(const bf16x8*)g;
#endif
}
__device__ __forceinline__ void async_wait() {
#if USE_ASYNC_LDS
  __builtin_amdgcn_s_wait_asynccnt(0);
#endif
}

__device__ __forceinline__ v16bf combine8(bf16x8 lo, bf16x8 hi) {
  v16bf r;
#pragma unroll
  for (int i = 0; i < 8; ++i) { r[i] = lo[i]; r[8 + i] = hi[i]; }
  return r;
}

// A-matrix fragment (16x32, 16-bit): per ISA layout, lane<16 holds K{0..7,16..23},
// lane>=16 holds K{8..15,24..31}; row = lane%16. base points at a row-major
// 16x32 tile with row stride `stride` (in elements).
__device__ __forceinline__ v16bf frag_a(const bf16_t* base, int stride, int lane) {
  const bf16_t* p = base + (size_t)(lane & 15) * stride + ((lane >> 4) << 3);
  bf16x8 lo = *(const bf16x8*)(p);
  bf16x8 hi = *(const bf16x8*)(p + 16);
  return combine8(lo, hi);
}

// B-matrix fragment (32x16, 16-bit): lane<16 holds K{0..7,16..23} of column
// N=lane%16, lane>=16 holds K{8..15,24..31} (same K order as frag_a).
// base points at a [n][k] tile with row (n) stride `stride`.
__device__ __forceinline__ v16bf frag_b(const bf16_t* base, int stride, int lane) {
  const bf16_t* p = base + (size_t)(lane & 15) * stride + ((lane >> 4) << 3);
  bf16x8 lo = *(const bf16x8*)(p);
  bf16x8 hi = *(const bf16x8*)(p + 16);
  return combine8(lo, hi);
}

__device__ __forceinline__ v8f wmma_bf16(v16bf a, v16bf b, v8f c) {
  return __builtin_amdgcn_wmma_f32_16x16x32_bf16(false, a, false, b,
                                                 (short)0, c, false, false);
}

// ---------------------------------------------------------------------------
// f32 -> bf16 conversion
// ---------------------------------------------------------------------------
__global__ void k_cvt(const float* __restrict__ in, bf16_t* __restrict__ out,
                      long long n) {
  long long i  = (long long)blockIdx.x * blockDim.x + threadIdx.x;
  long long st = (long long)gridDim.x * blockDim.x;
  for (; i < n; i += st) out[i] = f2bf(in[i]);
}

// ---------------------------------------------------------------------------
// LayerNorm: one block per row, cols == 1024, f32 in -> bf16 out
// ---------------------------------------------------------------------------
__global__ __launch_bounds__(256) void k_ln(const float* __restrict__ x,
                                            const float* __restrict__ w,
                                            const float* __restrict__ b,
                                            bf16_t* __restrict__ out, int cols) {
  __shared__ float red[256];
  const int row = blockIdx.x;
  const int tid = threadIdx.x;
  const float* xr = x + (size_t)row * cols;
  float vals[4];
  float s1 = 0.f, s2 = 0.f;
#pragma unroll
  for (int i = 0; i < 4; ++i) {
    float v = xr[tid + i * 256];
    vals[i] = v; s1 += v; s2 += v * v;
  }
  red[tid] = s1; __syncthreads();
  for (int st = 128; st > 0; st >>= 1) {
    if (tid < st) red[tid] += red[tid + st];
    __syncthreads();
  }
  float tot1 = red[0]; __syncthreads();
  red[tid] = s2; __syncthreads();
  for (int st = 128; st > 0; st >>= 1) {
    if (tid < st) red[tid] += red[tid + st];
    __syncthreads();
  }
  float tot2 = red[0];
  float mean = tot1 / (float)cols;
  float var  = tot2 / (float)cols - mean * mean;
  float rstd = rsqrtf(var + 1e-5f);
  bf16_t* orow = out + (size_t)row * cols;
#pragma unroll
  for (int i = 0; i < 4; ++i) {
    int c = tid + i * 256;
    orow[c] = f2bf((vals[i] - mean) * rstd * w[c] + b[c]);
  }
}

// ---------------------------------------------------------------------------
// GEGLU gate: out = a * gelu_exact(g)   (bf16 in/out)
// ---------------------------------------------------------------------------
__global__ void k_geglu(const bf16_t* __restrict__ a, const bf16_t* __restrict__ g,
                        bf16_t* __restrict__ out, long long n) {
  long long i  = (long long)blockIdx.x * blockDim.x + threadIdx.x;
  long long st = (long long)gridDim.x * blockDim.x;
  for (; i < n; i += st) {
    float gv = bf2f(g[i]);
    float av = bf2f(a[i]);
    float ge = 0.5f * gv * (1.0f + erff(gv * 0.70710678118654752f));
    out[i] = f2bf(av * ge);
  }
}

// ---------------------------------------------------------------------------
// Tiled bf16 WMMA GEMM: C[MxN] = A[MxK] * B[KxN] (+bias) (+residual)
// Block tile 128x128, BK=32, 256 threads = 8 waves, each wave 32x64.
// A tile staged via async global->LDS copies (ASYNCcnt); B tile transposed
// to [n][k] through VGPRs while the async engine fills A.
// ---------------------------------------------------------------------------
template <bool OUT_BF16, bool HAS_BIAS, bool HAS_RES>
__global__ __launch_bounds__(256) void k_gemm(
    const bf16_t* __restrict__ A, int lda,
    const bf16_t* __restrict__ B, int ldb,
    void* __restrict__ C, int ldc,
    const float* __restrict__ bias,
    const float* __restrict__ res, int ldr,
    int M, int N, int K) {
  __shared__ bf16_t As[128][32];   // [m][k]
  __shared__ bf16_t Bs[128][32];   // [n][k] (transposed on the way in)
  const int tid  = threadIdx.x;
  const int lane = tid & 31;
  const int wid  = tid >> 5;
  const int wm = wid & 3;          // 4 waves along M (32 rows each)
  const int wn = wid >> 2;         // 2 waves along N (64 cols each)
  const int m0 = blockIdx.y * 128;
  const int n0 = blockIdx.x * 128;

  v8f acc[2][4];
#pragma unroll
  for (int i = 0; i < 2; ++i)
#pragma unroll
    for (int j = 0; j < 4; ++j)
#pragma unroll
      for (int r = 0; r < 8; ++r) acc[i][j][r] = 0.f;

  const int arow = tid >> 1, akc = (tid & 1) << 4;   // A: 128 rows x 2 chunks
  const int bk   = tid >> 3, bnc = (tid & 7) << 4;   // B: 32 k-rows x 8 chunks

  // clamp OOB A rows to a valid address: their outputs are never stored and a
  // garbage A row only affects its own C row.
  int ar = m0 + arow; if (ar >= M) ar = M - 1;
  const bf16_t* gA = A + (size_t)ar * lda + akc;

  for (int k0 = 0; k0 < K; k0 += 32) {
    // kick off async A-tile fill (2 x 16B per thread)
    async_copy16(gA + k0,     &As[arow][akc]);
    async_copy16(gA + k0 + 8, &As[arow][akc + 8]);
    // meanwhile stage + transpose the B tile [k][n] -> [n][k]
    {
      const bf16_t* g = B + (size_t)(k0 + bk) * ldb + n0 + bnc;
      bf16x8 lo = *(const bf16x8*)g;
      bf16x8 hi = *(const bf16x8*)(g + 8);
#pragma unroll
      for (int j = 0; j < 8; ++j) {
        Bs[bnc + j][bk]     = lo[j];
        Bs[bnc + 8 + j][bk] = hi[j];
      }
      if (k0 + 32 < K)
        __builtin_prefetch(B + (size_t)(k0 + 32 + bk) * ldb + n0 + bnc, 0, 1);
    }
    async_wait();
    __syncthreads();

    v16bf af[2], bfr[4];
#pragma unroll
    for (int mi = 0; mi < 2; ++mi)
      af[mi] = frag_a(&As[wm * 32 + mi * 16][0], 32, lane);
#pragma unroll
    for (int ni = 0; ni < 4; ++ni)
      bfr[ni] = frag_b(&Bs[wn * 64 + ni * 16][0], 32, lane);
#pragma unroll
    for (int mi = 0; mi < 2; ++mi)
#pragma unroll
      for (int ni = 0; ni < 4; ++ni)
        acc[mi][ni] = wmma_bf16(af[mi], bfr[ni], acc[mi][ni]);
    __syncthreads();
  }

  // epilogue: C layout -> VGPR r holds row r (lanes 0-15) / r+8 (lanes 16-31)
  const int mhi = (lane >> 4) << 3;
  const int nlo = lane & 15;
#pragma unroll
  for (int mi = 0; mi < 2; ++mi) {
#pragma unroll
    for (int r = 0; r < 8; ++r) {
      int grow = m0 + wm * 32 + mi * 16 + r + mhi;
      if (grow >= M) continue;
#pragma unroll
      for (int ni = 0; ni < 4; ++ni) {
        int gcol = n0 + wn * 64 + ni * 16 + nlo;
        float v = acc[mi][ni][r];
        if (HAS_BIAS) v += bias[gcol];
        if (HAS_RES)  v += res[(size_t)grow * ldr + gcol];
        if (OUT_BF16) ((bf16_t*)C)[(size_t)grow * ldc + gcol] = f2bf(v);
        else          ((float*)C)[(size_t)grow * ldc + gcol]  = v;
      }
    }
  }
}

// ---------------------------------------------------------------------------
// Fused flash attention (bf16, d_head=64, heads laid out in cols of stride 1024)
// Block = 4 waves x 16 query rows = 64 queries, one head per blockIdx.y.
// Never materializes the NxN score matrix. K tiles go through the async
// global->LDS path (OOB keys clamped, then masked to -1e30); V keeps the
// zero-filled sync path so P@V never sees 0 x garbage.
// ---------------------------------------------------------------------------
__global__ __launch_bounds__(128) void k_flash(
    const bf16_t* __restrict__ Q, const bf16_t* __restrict__ K,
    const bf16_t* __restrict__ V, bf16_t* __restrict__ O,
    int nkv, float scale) {
  __shared__ bf16_t Ks[32][64];        // [key][d]
  __shared__ bf16_t Vs[64][32];        // [d][key] (transposed)
  __shared__ bf16_t Ps[4][16][32];     // per-wave P tile [row][key]
  const int tid  = threadIdx.x;
  const int lane = tid & 31;
  const int wid  = tid >> 5;
  const int h    = blockIdx.y;
  const int q0   = blockIdx.x * 64 + wid * 16;
  const size_t hoff = (size_t)h * 64;

  // Q fragments (16 rows x 64 d = two 16x32 A-frags), loaded once from global
  v16bf qf[2];
#pragma unroll
  for (int d = 0; d < 2; ++d)
    qf[d] = frag_a(Q + (size_t)q0 * 1024 + hoff + d * 32, 1024, lane);

  v8f o[4];
#pragma unroll
  for (int i = 0; i < 4; ++i)
#pragma unroll
    for (int r = 0; r < 8; ++r) o[i][r] = 0.f;
  float mrow[8], lrow[8];
#pragma unroll
  for (int r = 0; r < 8; ++r) { mrow[r] = -1e30f; lrow[r] = 0.f; }

  const int krow = tid >> 2, kc = (tid & 3) << 4;

  for (int kb = 0; kb < nkv; kb += 32) {
    {   // async K block (clamped; invalid keys masked after S)
      int kr = kb + krow; if (kr >= nkv) kr = nkv - 1;
      const bf16_t* gk = K + (size_t)kr * 1024 + hoff + kc;
      async_copy16(gk,     &Ks[krow][kc]);
      async_copy16(gk + 8, &Ks[krow][kc + 8]);
    }
    {   // sync V block, zero-filled, stored transposed
      bf16x8 vlo, vhi;
      if (kb + krow < nkv) {
        const bf16_t* gv = V + (size_t)(kb + krow) * 1024 + hoff + kc;
        vlo = *(const bf16x8*)gv;  vhi = *(const bf16x8*)(gv + 8);
      } else {
#pragma unroll
        for (int i = 0; i < 8; ++i) { vlo[i] = f2bf(0.f); vhi[i] = f2bf(0.f); }
      }
#pragma unroll
      for (int j = 0; j < 8; ++j) {
        Vs[kc + j][krow]     = vlo[j];
        Vs[kc + 8 + j][krow] = vhi[j];
      }
    }
    async_wait();
    __syncthreads();

    // S = Q @ K^T  (16 queries x 32 keys)
    v8f s[2];
#pragma unroll
    for (int kn = 0; kn < 2; ++kn)
#pragma unroll
      for (int r = 0; r < 8; ++r) s[kn][r] = 0.f;
#pragma unroll
    for (int d = 0; d < 2; ++d)
#pragma unroll
      for (int kn = 0; kn < 2; ++kn) {
        v16bf kfrag = frag_b(&Ks[kn * 16][d * 32], 64, lane);
        s[kn] = wmma_bf16(qf[d], kfrag, s[kn]);
      }

    // scale + key mask
#pragma unroll
    for (int kn = 0; kn < 2; ++kn) {
      int key = kb + kn * 16 + (lane & 15);
      bool valid = key < nkv;
#pragma unroll
      for (int r = 0; r < 8; ++r)
        s[kn][r] = valid ? s[kn][r] * scale : -1e30f;
    }

    // online softmax per row (rows spread over 16-lane halves; xor<=8 stays in-half)
#pragma unroll
    for (int r = 0; r < 8; ++r) {
      float mx = fmaxf(s[0][r], s[1][r]);
      mx = fmaxf(mx, __shfl_xor(mx, 1, 32));
      mx = fmaxf(mx, __shfl_xor(mx, 2, 32));
      mx = fmaxf(mx, __shfl_xor(mx, 4, 32));
      mx = fmaxf(mx, __shfl_xor(mx, 8, 32));
      float mnew = fmaxf(mrow[r], mx);
      float p0 = __expf(s[0][r] - mnew);
      float p1 = __expf(s[1][r] - mnew);
      float rs = p0 + p1;
      rs += __shfl_xor(rs, 1, 32);
      rs += __shfl_xor(rs, 2, 32);
      rs += __shfl_xor(rs, 4, 32);
      rs += __shfl_xor(rs, 8, 32);
      float sc = __expf(mrow[r] - mnew);
      lrow[r] = lrow[r] * sc + rs;
      mrow[r] = mnew;
#pragma unroll
      for (int ni = 0; ni < 4; ++ni) o[ni][r] *= sc;
      s[0][r] = p0; s[1][r] = p1;
    }

    // re-layout P through LDS: C-layout -> [row][key] for A-frag reuse
    {
      const int mhi = (lane >> 4) << 3, nlo = lane & 15;
#pragma unroll
      for (int kn = 0; kn < 2; ++kn)
#pragma unroll
        for (int r = 0; r < 8; ++r)
          Ps[wid][r + mhi][kn * 16 + nlo] = f2bf(s[kn][r]);
    }
    __syncthreads();

    // O += P @ V
    v16bf pf = frag_a(&Ps[wid][0][0], 32, lane);
#pragma unroll
    for (int ni = 0; ni < 4; ++ni) {
      v16bf vf = frag_b(&Vs[ni * 16][0], 32, lane);
      o[ni] = wmma_bf16(pf, vf, o[ni]);
    }
    __syncthreads();
  }

  // finalize: O /= l, store bf16
  const int mhi = (lane >> 4) << 3, nlo = lane & 15;
#pragma unroll
  for (int r = 0; r < 8; ++r) {
    float inv = 1.0f / lrow[r];
    int row = q0 + r + mhi;
#pragma unroll
    for (int ni = 0; ni < 4; ++ni)
      O[(size_t)row * 1024 + hoff + ni * 16 + nlo] = f2bf(o[ni][r] * inv);
  }
}

// ---------------------------------------------------------------------------
// Host orchestration
// ---------------------------------------------------------------------------
extern "C" void kernel_launch(void* const* d_in, const int* in_sizes, int n_in,
                              void* d_out, int out_size, void* d_ws, size_t ws_size,
                              hipStream_t stream) {
  (void)in_sizes; (void)n_in; (void)out_size; (void)ws_size;
  const float* x    = (const float*)d_in[0];
  const float* ctx  = (const float*)d_in[1];
  const float* n1w  = (const float*)d_in[2];
  const float* n1b  = (const float*)d_in[3];
  const float* n2w  = (const float*)d_in[4];
  const float* n2b  = (const float*)d_in[5];
  const float* n3w  = (const float*)d_in[6];
  const float* n3b  = (const float*)d_in[7];
  const float* q1w  = (const float*)d_in[8];
  const float* k1w  = (const float*)d_in[9];
  const float* v1w  = (const float*)d_in[10];
  const float* o1w  = (const float*)d_in[11];
  const float* o1b  = (const float*)d_in[12];
  const float* q2w  = (const float*)d_in[13];
  const float* k2w  = (const float*)d_in[14];
  const float* v2w  = (const float*)d_in[15];
  const float* o2w  = (const float*)d_in[16];
  const float* o2b  = (const float*)d_in[17];
  const float* ff1w = (const float*)d_in[18];
  const float* ff1b = (const float*)d_in[19];
  const float* ff2w = (const float*)d_in[20];
  const float* ff2b = (const float*)d_in[21];
  float* xout = (float*)d_out;

  char* wsp = (char*)d_ws;
  size_t off = 0;
  auto alloc = [&](size_t elems) -> bf16_t* {
    bf16_t* p = (bf16_t*)(wsp + off);
    off += ((elems * sizeof(bf16_t)) + 255) & ~(size_t)255;
    return p;
  };
  const size_t W11 = 1024UL * 1024, W21 = 768UL * 1024;
  const size_t WF1 = 1024UL * 8192, WF2 = 4096UL * 1024;
  bf16_t* bq1  = alloc(W11); bf16_t* bk1 = alloc(W11);
  bf16_t* bv1  = alloc(W11); bf16_t* bo1 = alloc(W11);
  bf16_t* bq2  = alloc(W11); bf16_t* bk2 = alloc(W21);
  bf16_t* bv2  = alloc(W21); bf16_t* bo2 = alloc(W11);
  bf16_t* bff1 = alloc(WF1); bf16_t* bff2 = alloc(WF2);
  bf16_t* bctx = alloc(77UL * 768);
  bf16_t* hb   = alloc(4096UL * 1024);
  bf16_t* qb   = alloc(4096UL * 1024);
  bf16_t* kb   = alloc(4096UL * 1024);
  bf16_t* vb   = alloc(4096UL * 1024);
  bf16_t* ab   = alloc(4096UL * 1024);   // attention output
  bf16_t* fa   = alloc(4096UL * 4096);   // GEGLU "a" half
  bf16_t* fg   = alloc(4096UL * 4096);   // GEGLU gate half
  bf16_t* fh   = alloc(4096UL * 4096);   // gated hidden

  auto cvt = [&](const float* src, bf16_t* dst, long long n) {
    int blocks = (int)((n + 1023) / 1024);
    if (blocks > 2048) blocks = 2048;
    k_cvt<<<blocks, 256, 0, stream>>>(src, dst, n);
  };
  auto gemm_bf = [&](const bf16_t* A, int lda, const bf16_t* B, int ldb,
                     bf16_t* C, int ldc, const float* bias, int M, int N, int K) {
    dim3 g(N / 128, (M + 127) / 128);
    if (bias)
      k_gemm<true, true, false><<<g, 256, 0, stream>>>(A, lda, B, ldb, C, ldc,
                                                       bias, nullptr, 0, M, N, K);
    else
      k_gemm<true, false, false><<<g, 256, 0, stream>>>(A, lda, B, ldb, C, ldc,
                                                        nullptr, nullptr, 0, M, N, K);
  };
  auto gemm_f32 = [&](const bf16_t* A, int lda, const bf16_t* B, int ldb,
                      float* C, int ldc, const float* bias, const float* res,
                      int M, int N, int K) {
    dim3 g(N / 128, (M + 127) / 128);
    k_gemm<false, true, true><<<g, 256, 0, stream>>>(A, lda, B, ldb, C, ldc,
                                                     bias, res, ldc, M, N, K);
  };

  // --- precision conversion (bf16 weights/context) ---
  cvt(q1w, bq1, (long long)W11); cvt(k1w, bk1, (long long)W11);
  cvt(v1w, bv1, (long long)W11); cvt(o1w, bo1, (long long)W11);
  cvt(q2w, bq2, (long long)W11); cvt(k2w, bk2, (long long)W21);
  cvt(v2w, bv2, (long long)W21); cvt(o2w, bo2, (long long)W11);
  cvt(ff1w, bff1, (long long)WF1); cvt(ff2w, bff2, (long long)WF2);
  cvt(ctx, bctx, 77LL * 768);

  const float scale = 0.125f;   // 64^-0.5

  // --- self-attention sublayer ---
  k_ln<<<4096, 256, 0, stream>>>(x, n1w, n1b, hb, 1024);
  gemm_bf(hb, 1024, bq1, 1024, qb, 1024, nullptr, 4096, 1024, 1024);
  gemm_bf(hb, 1024, bk1, 1024, kb, 1024, nullptr, 4096, 1024, 1024);
  gemm_bf(hb, 1024, bv1, 1024, vb, 1024, nullptr, 4096, 1024, 1024);
  k_flash<<<dim3(64, 16), 128, 0, stream>>>(qb, kb, vb, ab, 4096, scale);
  gemm_f32(ab, 1024, bo1, 1024, xout, 1024, o1b, x, 4096, 1024, 1024);

  // --- cross-attention sublayer ---
  k_ln<<<4096, 256, 0, stream>>>(xout, n2w, n2b, hb, 1024);
  gemm_bf(hb, 1024, bq2, 1024, qb, 1024, nullptr, 4096, 1024, 1024);
  gemm_bf(bctx, 768, bk2, 1024, kb, 1024, nullptr, 77, 1024, 768);
  gemm_bf(bctx, 768, bv2, 1024, vb, 1024, nullptr, 77, 1024, 768);
  k_flash<<<dim3(64, 16), 128, 0, stream>>>(qb, kb, vb, ab, 77, scale);
  gemm_f32(ab, 1024, bo2, 1024, xout, 1024, o2b, xout, 4096, 1024, 1024);

  // --- GEGLU feed-forward sublayer ---
  k_ln<<<4096, 256, 0, stream>>>(xout, n3w, n3b, hb, 1024);
  gemm_bf(hb, 1024, bff1, 8192, fa, 4096, ff1b, 4096, 4096, 1024);
  gemm_bf(hb, 1024, bff1 + 4096, 8192, fg, 4096, ff1b + 4096, 4096, 4096, 1024);
  k_geglu<<<2048, 256, 0, stream>>>(fa, fg, fh, 4096LL * 4096);
  gemm_f32(fh, 4096, bff2, 1024, xout, 1024, ff2b, xout, 4096, 1024, 4096);
}
